// CropHealthClassifier_86655260164620
// MI455X (gfx1250) — matrix operand order repaired and checked by reading, not compiled, not run on hardware
//
#include <hip/hip_runtime.h>

typedef _Float16 v8h  __attribute__((ext_vector_type(8)));
typedef _Float16 v16h __attribute__((ext_vector_type(16)));
typedef float    v8f  __attribute__((ext_vector_type(8)));

#define BB   4096
#define FF   32
#define TT   100
#define HH   512
#define CC   8
#define DD   256
#define BETA_F 0.95122945f   // exp(-1/20)
#define THR_F  1.0f

#define SHUF16(lo, hi) __builtin_shufflevector(lo, hi, 0,1,2,3,4,5,6,7,8,9,10,11,12,13,14,15)

// ---- prologue: convert W1 (H x D, f32) -> f16 copy in workspace ----
__global__ __launch_bounds__(256) void cvt_w1_kernel(const float* __restrict__ W1,
                                                     _Float16* __restrict__ W1h) {
  int i = blockIdx.x * 256 + threadIdx.x;
  if (i < HH * DD) W1h[i] = (_Float16)W1[i];
}

// ---- main SNN kernel: one workgroup per 16 batch rows, 16 waves ----
__global__ __launch_bounds__(512) void snn_lif_kernel(
    const float* __restrict__ spec,      // (B,F)
    const _Float16* __restrict__ W1h,    // (H,D) f16
    const float* __restrict__ W2,        // (C,H)
    const float* __restrict__ noise,     // (B,T,D)
    float* __restrict__ out)             // (B,C)
{
  __shared__ _Float16 spk[16 * DD];   // spike tile, row-major (16 x 256)
  __shared__ _Float16 s1l[16 * HH];   // layer-1 spikes (16 x 512)
  __shared__ _Float16 w2l[16 * HH];   // W2 padded to 16 rows (N) x 512 (K)

  const int tid  = threadIdx.x;
  const int w    = tid >> 5;          // wave id 0..15
  const int lane = tid & 31;
  const int lm   = lane & 15;
  const int lhi  = lane >> 4;         // 0 or 1
  const int b0   = blockIdx.x * 16;

  // ---- stage W2 into LDS, padded N 8->16 with zeros ----
  for (int i = tid; i < 16 * HH; i += 512) {
    int n = i >> 9, k = i & 511;
    w2l[i] = (n < CC) ? (_Float16)W2[n * HH + k] : (_Float16)0.0f;
  }

  // ---- population-encode probabilities: thread owns row w, d = f*8..f*8+7 ----
  const int row  = w;                  // batch row in tile
  const int f    = lane;               // feature 0..31
  const int dcol = f * 8;
  float p[8];
  {
    float s = spec[(b0 + row) * FF + f];
    #pragma unroll
    for (int j = 0; j < 8; ++j) {
      float d = s - (float)j * (1.0f / 7.0f);
      p[j] = 0.1f * __expf(-32.0f * d * d);   // 2*TW^2 = 1/32
    }
  }

  // ---- preload W1 B-fragments into registers (this wave's 32 hidden units) ----
  // B 32x16 f16, this code's contiguous k order (A uses the same): lanes 0-15 hold K=K0..K0+15 of column n=lane,
  //                     lanes 16-31 hold K=K0+16..K0+31 of column n=lane-16.
  v16h bfr[2][8];
  #pragma unroll
  for (int nt = 0; nt < 2; ++nt) {
    int h = w * 32 + nt * 16 + lm;
    #pragma unroll
    for (int k = 0; k < 8; ++k)
      bfr[nt][k] = *(const v16h*)(W1h + h * DD + k * 32 + lhi * 16);
  }

  v8f m1a0 = {}; v8f m1a1 = {};   // layer-1 membrane (C-layout, f32)
  v8f m2s  = {}; v8f accv = {};   // layer-2 membrane + spike accumulator (wave 0)

  __syncthreads();

  for (int t = 0; t < TT; ++t) {
    // ---- 1. encode spikes for this timestep into LDS ----
    {
      long base = ((long)(b0 + row) * TT + t) * DD + dcol;
      float4 u0 = *(const float4*)(noise + base);
      float4 u1 = *(const float4*)(noise + base + 4);
      v8h s;
      s[0] = (u0.x < p[0]) ? (_Float16)1.0f : (_Float16)0.0f;
      s[1] = (u0.y < p[1]) ? (_Float16)1.0f : (_Float16)0.0f;
      s[2] = (u0.z < p[2]) ? (_Float16)1.0f : (_Float16)0.0f;
      s[3] = (u0.w < p[3]) ? (_Float16)1.0f : (_Float16)0.0f;
      s[4] = (u1.x < p[4]) ? (_Float16)1.0f : (_Float16)0.0f;
      s[5] = (u1.y < p[5]) ? (_Float16)1.0f : (_Float16)0.0f;
      s[6] = (u1.z < p[6]) ? (_Float16)1.0f : (_Float16)0.0f;
      s[7] = (u1.w < p[7]) ? (_Float16)1.0f : (_Float16)0.0f;
      *(v8h*)(spk + row * DD + dcol) = s;
    }
    __syncthreads();

    // ---- 2. layer 1: m1 = beta*m1 + x @ W1^T via chained WMMA ----
    // Preload ALL 8 A-fragments first so the 16 ds_load_b128 pipeline
    // instead of stalling each WMMA pair on s_wait_dscnt 0.
    v16h afr[8];
    #pragma unroll
    for (int k = 0; k < 8; ++k) {
      // A 16x32 f16: lanes 0-15: K=lo..lo+15; lanes 16-31: +16
      const _Float16* ap = spk + lm * DD + k * 32 + lhi * 16;
      v8h lo = *(const v8h*)ap;
      v8h hi = *(const v8h*)(ap + 8);
      afr[k] = SHUF16(lo, hi);
    }
    v8f c0 = m1a0 * BETA_F;
    v8f c1 = m1a1 * BETA_F;
    #pragma unroll
    for (int k = 0; k < 8; ++k) {
      c0 = __builtin_amdgcn_wmma_f32_16x16x32_f16(false, afr[k], false, bfr[0][k],
                                                  (short)0, c0, false, false);
      c1 = __builtin_amdgcn_wmma_f32_16x16x32_f16(false, afr[k], false, bfr[1][k],
                                                  (short)0, c1, false, false);
    }
    // threshold + reset + publish s1 (C layout: elem i -> M = i + lhi*8, N = lm)
    #pragma unroll
    for (int i = 0; i < 8; ++i) {
      int m = i + lhi * 8;
      float v0 = c0[i]; bool f0 = v0 > THR_F;
      s1l[m * HH + w * 32 + lm]      = f0 ? (_Float16)1.0f : (_Float16)0.0f;
      c0[i] = f0 ? 0.0f : v0;
      float v1 = c1[i]; bool f1 = v1 > THR_F;
      s1l[m * HH + w * 32 + 16 + lm] = f1 ? (_Float16)1.0f : (_Float16)0.0f;
      c1[i] = f1 ? 0.0f : v1;
    }
    m1a0 = c0; m1a1 = c1;
    __syncthreads();

    // ---- 3. layer 2 (wave 0 only): m2 = beta*m2 + s1 @ W2^T ----
    if (w == 0) {
      v8f c2 = m2s * BETA_F;
      // batch fragment loads 4 k-steps at a time so ds loads amortize waits
      #pragma unroll
      for (int kk = 0; kk < 16; kk += 4) {
        v16h a2[4], b2[4];
        #pragma unroll
        for (int j = 0; j < 4; ++j) {
          int k = kk + j;
          const _Float16* ap = s1l + lm * HH + k * 32 + lhi * 16;
          v8h alo = *(const v8h*)ap;
          v8h ahi = *(const v8h*)(ap + 8);
          a2[j] = SHUF16(alo, ahi);
          const _Float16* bp = w2l + lm * HH + k * 32 + lhi * 16;
          v8h blo = *(const v8h*)bp;
          v8h bhi = *(const v8h*)(bp + 8);
          b2[j] = SHUF16(blo, bhi);
        }
        #pragma unroll
        for (int j = 0; j < 4; ++j)
          c2 = __builtin_amdgcn_wmma_f32_16x16x32_f16(false, a2[j], false, b2[j],
                                                      (short)0, c2, false, false);
      }
      #pragma unroll
      for (int i = 0; i < 8; ++i) {
        float v = c2[i]; bool fs = v > THR_F;
        accv[i] += fs ? 1.0f : 0.0f;
        c2[i] = fs ? 0.0f : v;
      }
      m2s = c2;
    }
  }

  // ---- output firing rates: out[b0+m][n] = acc/T (only n<8 valid) ----
  if (w == 0 && lm < CC) {
    #pragma unroll
    for (int i = 0; i < 8; ++i)
      out[(long)(b0 + i + lhi * 8) * CC + lm] = accv[i] * (1.0f / (float)TT);
  }
}

extern "C" void kernel_launch(void* const* d_in, const int* in_sizes, int n_in,
                              void* d_out, int out_size, void* d_ws, size_t ws_size,
                              hipStream_t stream) {
  const float* spec  = (const float*)d_in[0];   // (B,F)
  const float* W1    = (const float*)d_in[1];   // (H,D)
  const float* W2    = (const float*)d_in[2];   // (C,H)
  const float* noise = (const float*)d_in[3];   // (B,T,D)
  float* out = (float*)d_out;

  _Float16* W1h = (_Float16*)d_ws;              // 512*256*2 = 256 KB scratch

  cvt_w1_kernel<<<dim3((HH * DD + 255) / 256), dim3(256), 0, stream>>>(W1, W1h);
  snn_lif_kernel<<<dim3(BB / 16), dim3(512), 0, stream>>>(spec, W1h, W2, noise, out);
}
